// GenerativeSRNN_77610059039282
// MI455X (gfx1250) — matrix operand order repaired and checked by reading, not compiled, not run on hardware
//
#include <hip/hip_runtime.h>
#include <math.h>

// ---------------- types ----------------
typedef unsigned short u16;
typedef __attribute__((ext_vector_type(8)))  u16     u16x8;
typedef __attribute__((ext_vector_type(16))) __bf16  v16bf;
typedef __attribute__((ext_vector_type(8)))  float   v8f;
typedef __attribute__((ext_vector_type(4)))  unsigned su4;
typedef __attribute__((ext_vector_type(8)))  unsigned su8;

union V16 { u16x8 h[2]; v16bf v; };

// ---------------- problem dims ----------------
#define BB   32
#define TT   512
#define DD   64
#define KK   8
#define HH   256
#define TM1  511
#define MROWS (BB*TM1)          // 16352 = 1022 * 16

// log-domain constants: const = -0.5*dim*ln(2*pi*1e-3), inv = 1/(2*1e-4)
#define CONST_H 648.9444112093175f
#define CONST_Y 162.23610280232937f
#define INV2H   5000.0f
#define INV2Y   5000.0f

// ---------------- output layout (floats, concatenated) ----------------
#define OFF_P0 0
#define OFF_PS 256
#define OFF_PH 1048832
#define OFF_PY 1179904
#define OFF_G  1196288
#define OFF_DL 1327360
#define OFF_FW 2373888
#define OFF_BW 2504960

// ---------------- workspace layout (bytes) ----------------
#define WS_HB   0u          // bf16 h_samp  : 4,194,304 u16
#define WS_XB   8388608u    // bf16 x       : 1,048,576 u16
#define WS_WCAT 10485760u   // packed [Wih|Whh] B-tiles : 655,360 u16
#define WS_WT   11796480u   // packed trans_Wih B-tiles : 16,384 u16
#define WS_U    11829248u   // trans input proj : 1,046,528 f32
#define WS_LOGZ 16015360u   // 32 f32

// ---------------- helpers ----------------
__device__ __forceinline__ u16 f2bf(float f) {
  unsigned u = __builtin_bit_cast(unsigned, f);
  unsigned r = u + 0x7FFFu + ((u >> 16) & 1u);   // RNE
  return (u16)(r >> 16);
}

__device__ __forceinline__ float lane_bcast(float v, int srcLane) {
  int i = __builtin_bit_cast(int, v);
  i = __builtin_amdgcn_ds_bpermute(srcLane << 2, i);
  return __builtin_bit_cast(float, i);
}

template <int MASK>
__device__ __forceinline__ float xor_add(float v) {
  // ds_swizzle group-of-32: and=0x1F, or=0, xor=MASK
  int i = __builtin_amdgcn_ds_swizzle(__builtin_bit_cast(int, v), 0x1F | (MASK << 10));
  return v + __builtin_bit_cast(float, i);
}

__device__ __forceinline__ float lse8_ptr(const float* p) {
  float m = -1e30f;
  for (int q = 0; q < 8; ++q) m = fmaxf(m, p[q]);
  float s = 0.f;
  for (int q = 0; q < 8; ++q) s += expf(p[q] - m);
  return m + logf(s);
}

// ---------------- cast f32 -> bf16 ----------------
__global__ void __launch_bounds__(256) k_cast_bf16(const float* __restrict__ src,
                                                   u16* __restrict__ dst, int n) {
  int i = blockIdx.x * blockDim.x + threadIdx.x;
  if (i < n) dst[i] = f2bf(src[i]);
}

// ---------------- pack combined [Wih | Whh] into WMMA B-tile order ----------------
// buffer index = ((nt*10 + kc)*32 + lane)*16 + e ; n = nt*16 + (lane&15) ;
// K = kc*32 + (lane>>4)*8 + (e&7) + (e>>3)*16 ; K<64 -> Wih[d=K], K>=64 -> Whh[h=K-64]
__global__ void __launch_bounds__(256) k_pack_wcat(const float* __restrict__ Wih,
                                                   const float* __restrict__ Whh,
                                                   u16* __restrict__ out) {
  int idx = blockIdx.x * blockDim.x + threadIdx.x;
  if (idx >= 128 * 10 * 512) return;
  int e    = idx & 15;
  int L    = (idx >> 4) & 31;
  int rest = idx >> 9;              // nt*10 + kc
  int kc   = rest % 10;
  int nt   = rest / 10;
  int ncol = nt * 16 + (L & 15);    // (k_state, h_out)
  int Kd   = kc * 32 + ((L >> 4) * 8) + (e & 7) + ((e >> 3) * 16);
  int kst  = ncol >> 8, hc = ncol & 255;
  float v = (Kd < 64) ? Wih[(kst * HH + hc) * DD + Kd]
                      : Whh[(kst * HH + hc) * HH + (Kd - 64)];
  out[idx] = f2bf(v);
}

// ---------------- pack trans_Wih [64 x 256] into B-tile order ----------------
__global__ void __launch_bounds__(256) k_pack_wt(const float* __restrict__ Wt,
                                                 u16* __restrict__ out) {
  int idx = blockIdx.x * blockDim.x + threadIdx.x;
  if (idx >= 4 * 8 * 512) return;
  int e    = idx & 15;
  int L    = (idx >> 4) & 31;
  int rest = idx >> 9;              // nt*8 + kc
  int kc   = rest & 7;
  int nt   = rest >> 3;
  int j    = nt * 16 + (L & 15);
  int Kd   = kc * 32 + ((L >> 4) * 8) + (e & 7) + ((e >> 3) * 16);
  out[idx] = f2bf(Wt[j * HH + Kd]);
}

// ---------------- init: p_s[:,0]=eye, p_h[:,0], p0 ----------------
__global__ void __launch_bounds__(256) k_init(const float* __restrict__ h_samp,
                                              const float* __restrict__ pi0,
                                              float* __restrict__ out) {
  int idx = blockIdx.x * blockDim.x + threadIdx.x;
  if (idx < 2048) {                         // eye: b = idx/64, kk = idx%64
    int b = idx >> 6, kk = idx & 63;
    out[OFF_PS + b * TT * 64 + kk] = ((kk >> 3) == (kk & 7)) ? 1.f : 0.f;
  } else if (idx < 2304) {                  // p_h[b,0,k]
    int i = idx - 2048;
    int b = i >> 3, k = i & 7;
    float s = 0.f;
    const float* hr = h_samp + (b * TT) * HH;
    for (int h = 0; h < HH; ++h) { float v = hr[h]; s += v * v; }
    out[OFF_PH + (b * TT) * KK + k] = CONST_H - INV2H * s;
  } else if (idx < 2560) {                  // p0[b,k] = pi0 - lse(pi0)
    int i = idx - 2304;
    float lse = lse8_ptr(pi0);
    out[OFF_P0 + i] = pi0[i & 7] - lse;
  }
}

// ---------------- WMMA GEMM: u = h_samp[:, :-1] @ trans_Wih^T + bih + bhh ----------------
__global__ void __launch_bounds__(128) k_gemm_u(const u16* __restrict__ hb,
                                                const u16* __restrict__ wt,
                                                const float* __restrict__ bih,
                                                const float* __restrict__ bhh,
                                                float* __restrict__ u) {
  int w = blockIdx.x * 4 + (threadIdx.x >> 5);
  if (w >= MROWS / 16) return;
  int lane = threadIdx.x & 31;
  int mt = w;
  int row = mt * 16 + (lane & 15);
  int b = row / TM1, tp = row % TM1;
  const u16* arow = hb + (b * TT + tp) * HH;
  int hoff = (lane >> 4) * 8;

  V16 a[8];
  for (int kc = 0; kc < 8; ++kc) {
    const u16* p = arow + kc * 32 + hoff;
    a[kc].h[0] = *(const u16x8*)p;
    a[kc].h[1] = *(const u16x8*)(p + 16);
  }
  int mrow[8];
  for (int r = 0; r < 8; ++r) mrow[r] = mt * 16 + (lane >> 4) * 8 + r;

  for (int nt = 0; nt < 4; ++nt) {
    v8f acc = {};
    for (int kc = 0; kc < 8; ++kc) {
      V16 bm;
      const u16* bp = wt + ((nt * 8 + kc) * 32 + lane) * 16;
      bm.h[0] = *(const u16x8*)bp;
      bm.h[1] = *(const u16x8*)(bp + 8);
      acc = __builtin_amdgcn_wmma_f32_16x16x32_bf16(false, a[kc].v, false, bm.v,
                                                    (short)0, acc, false, false);
    }
    int j = nt * 16 + (lane & 15);
    float bias = bih[j] + bhh[j];
    for (int r = 0; r < 8; ++r) u[mrow[r] * 64 + j] = acc[r] + bias;
  }
}

// ---------------- WMMA GEMM + fused tanh/diff^2: p_h[:,1:] ----------------
// one workgroup (8 waves) per 16-row M-tile; wave wv handles k_state = wv.
// A tile staged once into LDS via GLOBAL_LOAD_ASYNC_TO_LDS_B128 (ASYNCcnt).
__global__ void __launch_bounds__(256) k_gemm_pre(const u16* __restrict__ hb,
                                                  const u16* __restrict__ xb,
                                                  const u16* __restrict__ wcat,
                                                  const float* __restrict__ bih,
                                                  const float* __restrict__ bhh,
                                                  const float* __restrict__ h_samp,
                                                  float* __restrict__ out) {
  __shared__ u16 sa[16 * 320];                   // A tile, combined-K layout [x | h]
  int tid  = threadIdx.x;
  int mt   = blockIdx.x;                         // 0..1021
  int lane = tid & 31;
  int ks   = tid >> 5;                           // wave id = k_state

  // ---- async-stage A tile: 640 units of 16B, per-lane gather into LDS ----
  unsigned sbase = (unsigned)(uintptr_t)(&sa[0]);
  for (int i = tid; i < 640; i += 256) {
    int r = i / 40, c = i % 40;                  // 40 x 16B units per row
    int m = mt * 16 + r;
    int br = m / TM1, tr = m % TM1;
    const u16* src = (c < 8) ? (xb + (br * TT + tr + 1) * DD + c * 8)
                             : (hb + (br * TT + tr) * HH + (c * 8 - 64));
    unsigned long long ga = (unsigned long long)(uintptr_t)src;
    unsigned ld = sbase + (unsigned)i * 16u;
    asm volatile("global_load_async_to_lds_b128 %0, %1, off"
                 :: "v"(ld), "v"(ga) : "memory");
  }
  asm volatile("s_wait_asynccnt 0x0" ::: "memory");
  __syncthreads();

  // ---- read hoisted A fragments from LDS (ds_load_b128 pairs) ----
  int hoff = (lane >> 4) * 8;
  const u16* arow = sa + (lane & 15) * 320;
  V16 a[10];
  for (int kc = 0; kc < 10; ++kc) {
    const u16* p = arow + kc * 32 + hoff;
    a[kc].h[0] = *(const u16x8*)p;
    a[kc].h[1] = *(const u16x8*)(p + 16);
  }

  int tgt[8], phx[8];
  for (int r = 0; r < 8; ++r) {
    int m  = mt * 16 + (lane >> 4) * 8 + r;
    int br = m / TM1, tr = m % TM1;
    tgt[r] = (br * TT + tr + 1) * HH;            // h_samp[b, t'+1, :]
    phx[r] = (br * TT + tr + 1) * KK + ks;       // p_h[b, t'+1, ks]
  }

  float ss[8];
  for (int r = 0; r < 8; ++r) ss[r] = 0.f;
  int ncl = lane & 15;

  for (int ntl = 0; ntl < 16; ++ntl) {
    int ntg = ks * 16 + ntl;
    const u16* bbase = wcat + ntg * 10 * 512 + lane * 16;
    __builtin_prefetch(bbase + 10 * 512, 0, 1);  // global_prefetch next n-tile
    v8f acc = {};
    V16 bm, bn;
    bm.h[0] = *(const u16x8*)bbase;
    bm.h[1] = *(const u16x8*)(bbase + 8);
    for (int kc = 0; kc < 10; ++kc) {
      if (kc < 9) {                              // double-buffer next B fragment
        const u16* np = bbase + (kc + 1) * 512;
        bn.h[0] = *(const u16x8*)np;
        bn.h[1] = *(const u16x8*)(np + 8);
      }
      acc = __builtin_amdgcn_wmma_f32_16x16x32_bf16(false, a[kc].v, false, bm.v,
                                                    (short)0, acc, false, false);
      if (kc < 9) bm = bn;
    }
    int hout = ntl * 16 + ncl;
    float bias = bih[ks * HH + hout] + bhh[ks * HH + hout];
    for (int r = 0; r < 8; ++r) {
      float mh = tanhf(acc[r] + bias);
      float d  = h_samp[tgt[r] + hout] - mh;
      ss[r] += d * d;
    }
  }
  for (int r = 0; r < 8; ++r) {                  // sum over 16 lanes of same half
    float s = ss[r];
    s = xor_add<1>(s); s = xor_add<2>(s); s = xor_add<4>(s); s = xor_add<8>(s);
    ss[r] = s;
  }
  if ((lane & 15) == 0)
    for (int r = 0; r < 8; ++r)
      out[OFF_PH + phx[r]] = CONST_H - INV2H * ss[r];
}

// ---------------- sequential transition RNN + row logsumexp -> p_s[:,1:] ----------------
// Whh (16KB contiguous) DMA'd into LDS by the Tensor Data Mover.
__global__ void __launch_bounds__(32) k_trans_seq(const float* __restrict__ u,
                                                  const float* __restrict__ Whh,
                                                  float* __restrict__ out) {
  __shared__ float wsh[64 * 64];
  __shared__ float fb[2][64];
  int b = blockIdx.x, lane = threadIdx.x;

  // ---- TDM descriptor: 1-row tile of 4096 x 4B elements ----
  unsigned long long ga = (unsigned long long)(uintptr_t)Whh;
  unsigned ldsa = (unsigned)(uintptr_t)(&wsh[0]);
  su4 g0;
  g0[0] = 1u;                                        // count=1, user mode
  g0[1] = ldsa;                                      // lds_addr (bits 63:32)
  g0[2] = (unsigned)ga;                              // global_addr[31:0]
  g0[3] = ((unsigned)(ga >> 32) & 0x01FFFFFFu)       // global_addr[56:32]
          | 0x80000000u;                             // type=2 ("image")
  su8 g1;
  g1[0] = 2u << 16;                                  // data_size = 4 bytes
  g1[1] = 4096u << 16;                               // tensor_dim0 = 4096 (lo16)
  g1[2] = 1u << 16;                                  // dim0 hi=0 | tensor_dim1 = 1
  g1[3] = 4096u << 16;                               // dim1 hi=0 | tile_dim0 = 4096
  g1[4] = 1u;                                        // tile_dim1 = 1, tile_dim2 = 0
  g1[5] = 4096u;                                     // tensor_dim0_stride = 4096
  g1[6] = 0u;
  g1[7] = 0u;
  asm volatile("tensor_load_to_lds %0, %1" :: "s"(g0), "s"(g1) : "memory");
  __builtin_amdgcn_s_wait_tensorcnt(0);

  fb[0][lane] = 0.f; fb[0][lane + 32] = 0.f;
  __syncthreads();
  for (int s = 0; s < TM1; ++s) {
    const float* up  = u + (b * TM1 + s) * 64;
    float* prev = fb[s & 1];
    float* cur  = fb[(s & 1) ^ 1];
    for (int jj = 0; jj < 2; ++jj) {
      int j = lane + jj * 32;
      float acc = up[j];
      const float* wr = wsh + j * 64;
      for (int i = 0; i < 64; ++i) acc += prev[i] * wr[i];
      cur[j] = tanhf(acc);
    }
    __syncthreads();
    float* psrow = out + OFF_PS + (b * TT + s + 1) * 64;
    for (int jj = 0; jj < 2; ++jj) {
      int j = lane + jj * 32;
      float lse = lse8_ptr(cur + (j >> 3) * 8);
      psrow[j] = cur[j] - lse;
    }
    __syncthreads();
  }
}

// ---------------- emission MLP -> p_y ----------------
__global__ void __launch_bounds__(256) k_emis(const float* __restrict__ h_samp,
                                              const float* __restrict__ y,
                                              const float* __restrict__ W1, const float* __restrict__ b1,
                                              const float* __restrict__ W2, const float* __restrict__ b2,
                                              const float* __restrict__ W3, const float* __restrict__ b3,
                                              float* __restrict__ out) {
  __shared__ float sm[8][HH + 32 + 64];
  int wv = threadIdx.x >> 5, lane = threadIdx.x & 31;
  int site = blockIdx.x * 8 + wv;                // 0..16383
  float* hs = sm[wv];
  float* h1 = hs + HH;
  float* h2 = h1 + 32;
  const float* hr = h_samp + (long long)site * HH;
  for (int i = 0; i < 8; ++i) hs[lane * 8 + i] = hr[lane * 8 + i];
  __syncthreads();
  {
    float acc = b1[lane];
    const float* wr = W1 + lane * HH;
    for (int h = 0; h < HH; ++h) acc += hs[h] * wr[h];
    h1[lane] = fmaxf(acc, 0.f);
  }
  __syncthreads();
  for (int jj = 0; jj < 2; ++jj) {
    int j = lane + jj * 32;
    float acc = b2[j];
    const float* wr = W2 + j * 32;
    for (int i = 0; i < 32; ++i) acc += h1[i] * wr[i];
    h2[j] = fmaxf(acc, 0.f);
  }
  __syncthreads();
  float local = 0.f;
  const float* yr = y + (long long)site * DD;
  for (int jj = 0; jj < 2; ++jj) {
    int j = lane + jj * 32;
    float acc = b3[j];
    const float* wr = W3 + j * 64;
    for (int i = 0; i < 64; ++i) acc += h2[i] * wr[i];
    float d = yr[j] - acc;
    local += d * d;
  }
  local = xor_add<16>(local); local = xor_add<8>(local); local = xor_add<4>(local);
  local = xor_add<2>(local);  local = xor_add<1>(local);
  if (lane == 0) out[OFF_PY + site] = CONST_Y - INV2Y * local;
}

// ---------------- HMM forward (alpha), LDS-staged ----------------
__global__ void __launch_bounds__(256) k_fwd(const float* __restrict__ pi0,
                                             float* __restrict__ out) {
  extern __shared__ float smem[];
  float* ps_l = smem;                 // [511][64] = p_s[b, 1:, :, :]
  float* ll_l = smem + TM1 * 64;      // [512][8]
  int b = blockIdx.x, tid = threadIdx.x;
  const float* psg = out + OFF_PS + b * TT * 64 + 64;
  for (int i = tid; i < TM1 * 64; i += 256) ps_l[i] = psg[i];
  for (int i = tid; i < TT * KK; i += 256) {
    int t = i >> 3, k = i & 7;
    ll_l[i] = out[OFF_PH + (b * TT + t) * KK + k] + out[OFF_PY + b * TT + t];
  }
  __syncthreads();
  if (tid < 8) {
    int k = tid;
    float lsep = lse8_ptr(pi0);
    float aa = (pi0[k] - lsep) + ll_l[k];
    out[OFF_FW + (b * TT) * KK + k] = aa;
    for (int t = 1; t < TT; ++t) {
      const float* pr = ps_l + (t - 1) * 64;
      float vv[8], m = -1e30f;
      for (int k1 = 0; k1 < 8; ++k1) {
        vv[k1] = lane_bcast(aa, k1) + pr[k1 * 8 + k];
        m = fmaxf(m, vv[k1]);
      }
      float s = 0.f;
      for (int k1 = 0; k1 < 8; ++k1) s += expf(vv[k1] - m);
      aa = m + logf(s) + ll_l[t * 8 + k];
      out[OFF_FW + (b * TT + t) * KK + k] = aa;
    }
  }
}

// ---------------- HMM backward (beta), LDS-staged ----------------
__global__ void __launch_bounds__(256) k_bwd(float* __restrict__ out) {
  extern __shared__ float smem[];
  float* ps_l = smem;
  float* ll_l = smem + TM1 * 64;
  int b = blockIdx.x, tid = threadIdx.x;
  const float* psg = out + OFF_PS + b * TT * 64 + 64;
  for (int i = tid; i < TM1 * 64; i += 256) ps_l[i] = psg[i];
  for (int i = tid; i < TT * KK; i += 256) {
    int t = i >> 3, k = i & 7;
    ll_l[i] = out[OFF_PH + (b * TT + t) * KK + k] + out[OFF_PY + b * TT + t];
  }
  __syncthreads();
  if (tid < 8) {
    int k1 = tid;
    float bb = 0.f;
    out[OFF_BW + (b * TT + TM1) * KK + k1] = 0.f;
    for (int t = TM1; t >= 1; --t) {
      const float* pr = ps_l + (t - 1) * 64 + k1 * 8;
      float sval = ll_l[t * 8 + k1] + bb;        // this lane's ll[t,k]+b[k]
      float vv[8], m = -1e30f;
      for (int k2 = 0; k2 < 8; ++k2) {
        vv[k2] = pr[k2] + lane_bcast(sval, k2);
        m = fmaxf(m, vv[k2]);
      }
      float s = 0.f;
      for (int k2 = 0; k2 < 8; ++k2) s += expf(vv[k2] - m);
      bb = m + logf(s);
      out[OFF_BW + (b * TT + t - 1) * KK + k1] = bb;
    }
  }
}

// ---------------- logZ ----------------
__global__ void __launch_bounds__(32) k_logz(const float* __restrict__ out,
                                             float* __restrict__ logZ) {
  int b = threadIdx.x;
  if (b < BB) logZ[b] = lse8_ptr(out + OFF_FW + (b * TT + TM1) * KK);
}

// ---------------- gamma & delta ----------------
__global__ void __launch_bounds__(256) k_gamma_delta(float* __restrict__ out,
                                                     const float* __restrict__ logZ) {
  int idx = blockIdx.x * blockDim.x + threadIdx.x;
  const int NG = BB * TT * KK;           // 131072
  const int ND = BB * TM1 * 64;          // 1046528
  if (idx < NG) {
    int b = idx / (TT * KK);
    out[OFF_G + idx] = expf(out[OFF_FW + idx] + out[OFF_BW + idx] - logZ[b]);
  } else if (idx < NG + ND) {
    int j  = idx - NG;
    int k2 = j & 7;
    int k1 = (j >> 3) & 7;
    int tp = (j >> 6) % TM1;
    int b  = j / (TM1 * 64);
    float llv = out[OFF_PH + (b * TT + tp + 1) * KK + k2] + out[OFF_PY + b * TT + tp + 1];
    float v = out[OFF_FW + (b * TT + tp) * KK + k1]
            + out[OFF_PS + (b * TT + tp + 1) * 64 + k1 * 8 + k2]
            + llv
            + out[OFF_BW + (b * TT + tp + 1) * KK + k2]
            - logZ[b];
    out[OFF_DL + j] = expf(v);
  }
}

// ---------------- launch ----------------
extern "C" void kernel_launch(void* const* d_in, const int* in_sizes, int n_in,
                              void* d_out, int out_size, void* d_ws, size_t ws_size,
                              hipStream_t stream) {
  const float* x      = (const float*)d_in[0];
  const float* y      = (const float*)d_in[1];
  const float* h_samp = (const float*)d_in[2];
  const float* pi0    = (const float*)d_in[3];
  const float* rWih   = (const float*)d_in[4];
  const float* rWhh   = (const float*)d_in[5];
  const float* rbih   = (const float*)d_in[6];
  const float* rbhh   = (const float*)d_in[7];
  const float* tWih   = (const float*)d_in[8];
  const float* tWhh   = (const float*)d_in[9];
  const float* tbih   = (const float*)d_in[10];
  const float* tbhh   = (const float*)d_in[11];
  const float* W1 = (const float*)d_in[12];
  const float* b1 = (const float*)d_in[13];
  const float* W2 = (const float*)d_in[14];
  const float* b2 = (const float*)d_in[15];
  const float* W3 = (const float*)d_in[16];
  const float* b3 = (const float*)d_in[17];
  float* out = (float*)d_out;
  char* ws = (char*)d_ws;
  u16*   hb   = (u16*)(ws + WS_HB);
  u16*   xb   = (u16*)(ws + WS_XB);
  u16*   wcat = (u16*)(ws + WS_WCAT);
  u16*   wt   = (u16*)(ws + WS_WT);
  float* u    = (float*)(ws + WS_U);
  float* logZ = (float*)(ws + WS_LOGZ);

  k_cast_bf16<<<16384, 256, 0, stream>>>(h_samp, hb, BB * TT * HH);
  k_cast_bf16<<<4096, 256, 0, stream>>>(x, xb, BB * TT * DD);
  k_pack_wcat<<<2560, 256, 0, stream>>>(rWih, rWhh, wcat);
  k_pack_wt<<<64, 256, 0, stream>>>(tWih, wt);
  k_init<<<10, 256, 0, stream>>>(h_samp, pi0, out);

  k_gemm_u<<<256, 128, 0, stream>>>(hb, wt, tbih, tbhh, u);       // 1022 waves
  k_trans_seq<<<32, 32, 0, stream>>>(u, tWhh, out);
  k_gemm_pre<<<1022, 256, 0, stream>>>(hb, xb, wcat, rbih, rbhh, h_samp, out); // 8176 waves
  k_emis<<<2048, 256, 0, stream>>>(h_samp, y, W1, b1, W2, b2, W3, b3, out);

  size_t lds = (size_t)(TM1 * 64 + TT * KK) * sizeof(float);       // 147,200 B (CDNA5 320KB LDS)
  k_fwd<<<32, 256, lds, stream>>>(pi0, out);
  k_bwd<<<32, 256, lds, stream>>>(out);
  k_logz<<<1, 32, 0, stream>>>(out, logZ);
  k_gamma_delta<<<4600, 256, 0, stream>>>(out, logZ);

  (void)in_sizes; (void)n_in; (void)out_size; (void)ws_size; (void)tWhh;
}
